// PromptAttention_14224931684857
// MI455X (gfx1250) — matrix operand order repaired and checked by reading, not compiled, not run on hardware
//
#include <hip/hip_runtime.h>
#include <cstdint>

// ---------------------------------------------------------------------------
// CDNA5 (gfx1250) prompt-attention: all three GEMM stages on
// v_wmma_f32_16x16x32_bf16; attention k/v chunks staged to LDS with
// double-buffered global_load_async_to_lds_b128 (ASYNCcnt); flash-softmax f32.
// ---------------------------------------------------------------------------

typedef __attribute__((ext_vector_type(16))) __bf16 v16bf;
typedef __attribute__((ext_vector_type(8)))  __bf16 v8bf;
typedef __attribute__((ext_vector_type(8)))  float  v8f;

union V16 { v16bf v; v8bf h[2]; };

constexpr int   DIM      = 768;
constexpr int   HEADS    = 12;
constexpr int   DH       = 64;
constexpr int   PROMPT   = 8;
constexpr int   INNER    = 768;
constexpr int   BB       = 4;
constexpr int   NQ       = 2048;
constexpr int   MROWS    = BB * NQ;        // 8192
constexpr int   NKREAL   = NQ + PROMPT;    // 2056
constexpr int   NKPAD    = 2080;           // next multiple of 32
constexpr int   QKV_COLS = 3 * INNER;      // 2304
constexpr float SCALE    = 0.125f;         // 64^-0.5

__device__ __forceinline__ v8f wmma_bf16(v16bf a, v16bf b, v8f c) {
    return __builtin_amdgcn_wmma_f32_16x16x32_bf16(
        /*neg_a=*/false, a, /*neg_b=*/false, b,
        /*c_mod=*/(short)0, c, /*reuse_a=*/false, /*reuse_b=*/false);
}

// Per-lane 16-byte async copy global -> LDS (tracked by ASYNCcnt).
__device__ __forceinline__ void async_fill_b128(unsigned lds_off,
                                                const __bf16* gaddr) {
    asm volatile("global_load_async_to_lds_b128 %0, %1, off"
                 :: "v"(lds_off), "v"((unsigned long long)(uintptr_t)gaddr)
                 : "memory");
}

// ---------------------------------------------------------------------------
// fp32 -> bf16 converters
// ---------------------------------------------------------------------------
__global__ void cvt_copy_kernel(const float* __restrict__ src,
                                __bf16* __restrict__ dst, int n) {
    for (int i = blockIdx.x * blockDim.x + threadIdx.x; i < n;
         i += gridDim.x * blockDim.x)
        dst[i] = (__bf16)src[i];
}

// src [rows][cols] fp32 -> dst [cols][rows] bf16
__global__ void cvt_transpose_kernel(const float* __restrict__ src,
                                     __bf16* __restrict__ dst,
                                     int rows, int cols) {
    int total = rows * cols;
    for (int i = blockIdx.x * blockDim.x + threadIdx.x; i < total;
         i += gridDim.x * blockDim.x) {
        int r = i / cols, c = i - r * cols;
        dst[(size_t)c * rows + r] = (__bf16)src[i];
    }
}

// ---------------------------------------------------------------------------
// QKV GEMM: xb[8192][768] x Wt[2304][768] (bf16, pre-transposed).
// Epilogue scatters into head-major q (scaled), k, and transposed v.
// ---------------------------------------------------------------------------
__global__ __launch_bounds__(256)
void qkv_gemm_kernel(const __bf16* __restrict__ xb,
                     const __bf16* __restrict__ wt,
                     __bf16* __restrict__ qs,
                     __bf16* __restrict__ kb,
                     __bf16* __restrict__ vt) {
    const int wave = threadIdx.x >> 5;
    const int lane = threadIdx.x & 31;
    const int ln   = lane & 15;
    const int lh   = lane >> 4;
    const int m0   = blockIdx.y * 128 + wave * 16;
    const int n0   = blockIdx.x * 64;

    v8f acc[4] = {};
    const __bf16* xr = xb + (size_t)(m0 + ln) * DIM;
    for (int kk = 0; kk < DIM; kk += 32) {
        if (kk + 32 < DIM) __builtin_prefetch(xr + kk + 32, 0, 0);
        V16 a;
        a.h[0] = *(const v8bf*)(xr + kk + lh * 8);
        a.h[1] = *(const v8bf*)(xr + kk + 16 + lh * 8);
        for (int nt = 0; nt < 4; ++nt) {
            const __bf16* wr = wt + (size_t)(n0 + nt * 16 + ln) * DIM + kk + lh * 8;
            V16 b;
            b.h[0] = *(const v8bf*)(wr);
            b.h[1] = *(const v8bf*)(wr + 16);
            acc[nt] = wmma_bf16(a.v, b.v, acc[nt]);
        }
    }
    for (int nt = 0; nt < 4; ++nt) {
        int col = n0 + nt * 16 + ln;      // 0..2303
        int sec = col / INNER;            // 0=q 1=k 2=v (uniform per tile)
        int cc  = col - sec * INNER;
        int h   = cc / DH, d = cc - (cc / DH) * DH;
        for (int j = 0; j < 8; ++j) {
            int m  = m0 + j + 8 * lh;
            int b  = m >> 11, nn = m & (NQ - 1);
            size_t bh = (size_t)b * HEADS + h;
            float val = acc[nt][j];
            if (sec == 0)
                qs[(bh * NQ + nn) * DH + d] = (__bf16)(val * SCALE);
            else if (sec == 1)
                kb[(bh * NKPAD + nn) * DH + d] = (__bf16)val;
            else
                vt[(bh * DH + d) * NKPAD + nn] = (__bf16)val;
        }
    }
}

// ---------------------------------------------------------------------------
// Prompt rows (n = 2048..2055) + zero pad (2056..2079); grid = B*H blocks.
// ---------------------------------------------------------------------------
__global__ void prompt_fill_kernel(const float* __restrict__ pk,
                                   const float* __restrict__ pv,
                                   __bf16* __restrict__ kb,
                                   __bf16* __restrict__ vt) {
    const int bh = blockIdx.x;
    const int h  = bh % HEADS;
    const int tail = NKPAD - NQ;          // 32
    for (int idx = threadIdx.x; idx < tail * DH; idx += blockDim.x) {
        int r = idx / DH, d = idx - r * DH;
        int n = NQ + r;
        float vk = 0.f, vv = 0.f;
        if (r < PROMPT) {
            vk = pk[((size_t)h * PROMPT + r) * DH + d];
            vv = pv[((size_t)h * PROMPT + r) * DH + d];
        }
        kb[((size_t)bh * NKPAD + n) * DH + d] = (__bf16)vk;
        vt[((size_t)bh * DH + d) * NKPAD + n] = (__bf16)vv;
    }
}

// ---------------------------------------------------------------------------
// Flash attention. grid (B*H, NQ/128), block = 256 thr = 8 waves.
// Each wave owns 16 query rows; k/v chunks double-buffered in LDS via
// async-to-LDS copies shared by all 8 waves.
// ---------------------------------------------------------------------------
__global__ __launch_bounds__(256)
void attn_kernel(const __bf16* __restrict__ qs,
                 const __bf16* __restrict__ kb,
                 const __bf16* __restrict__ vt,
                 __bf16* __restrict__ attn_out) {
    const int bh   = blockIdx.x;
    const int b    = bh / HEADS, h = bh % HEADS;
    const int tid  = threadIdx.x;
    const int wave = tid >> 5;
    const int lane = tid & 31;
    const int ln   = lane & 15;
    const int lh   = lane >> 4;
    const int mbase = blockIdx.y * 128 + wave * 16;

    const __bf16* qrow = qs + (size_t)bh * NQ * DH;
    const __bf16* krow = kb + (size_t)bh * NKPAD * DH;
    const __bf16* vrow = vt + (size_t)bh * DH * NKPAD;

    __shared__ __align__(16) __bf16 kbufs[2][32][64];   // [key][d]
    __shared__ __align__(16) __bf16 vbufs[2][64][32];   // [d][key]
    __shared__ __align__(16) __bf16 plds[8][16][32];    // per-wave P tile

    // q A-fragments (K = 0..31 and 32..63), loaded once, reused for all keys
    V16 aq[2];
    {
        const __bf16* qr = qrow + (size_t)(mbase + ln) * DH;
        for (int g = 0; g < 2; ++g) {
            aq[g].h[0] = *(const v8bf*)(qr + g * 32 + lh * 8);
            aq[g].h[1] = *(const v8bf*)(qr + g * 32 + 16 + lh * 8);
        }
    }

    v8f   acc[4] = {};
    float mstat[8], lstat[8];
    for (int j = 0; j < 8; ++j) { mstat[j] = -3.0e38f; lstat[j] = 0.f; }

    // Per-thread fill assignment: k chunk is one contiguous 4 KB span,
    // v chunk is 64 rows x 64 B from the transposed layout.
    const int vd = tid >> 2, vq = tid & 3;
    auto fill = [&](int buf, int kc) {
        const int keybase = kc * 32;
        async_fill_b128((unsigned)(uintptr_t)(&kbufs[buf][0][0] + tid * 8),
                        krow + (size_t)keybase * DH + tid * 8);
        async_fill_b128((unsigned)(uintptr_t)(&vbufs[buf][vd][vq * 8]),
                        vrow + (size_t)vd * NKPAD + keybase + vq * 8);
    };

    constexpr int NCH = NKPAD / 32;   // 65 chunks of 32 keys
    fill(0, 0);
    for (int kc = 0; kc < NCH; ++kc) {
        const int cur = kc & 1;
        const int keybase = kc * 32;
        if (kc + 1 < NCH) {
            fill(cur ^ 1, kc + 1);    // overlap next fill with this compute
            asm volatile("s_wait_asynccnt 0x2" ::: "memory");
        } else {
            asm volatile("s_wait_asynccnt 0x0" ::: "memory");
        }
        __syncthreads();              // current buffer visible to all waves

        // ---- S = q . k^T  (two 16x16 score tiles) --------------------------
        v8f st[2];
        for (int t = 0; t < 2; ++t) {
            const __bf16* kp = &kbufs[cur][t * 16 + ln][lh * 8];
            V16 bk0, bk1;
            bk0.h[0] = *(const v8bf*)(kp);
            bk0.h[1] = *(const v8bf*)(kp + 16);
            bk1.h[0] = *(const v8bf*)(kp + 32);
            bk1.h[1] = *(const v8bf*)(kp + 48);
            v8f s = {};
            s = wmma_bf16(aq[0].v, bk0.v, s);
            s = wmma_bf16(aq[1].v, bk1.v, s);
            int col = keybase + t * 16 + ln;
            if (col >= NKREAL)               // mask pad keys
                for (int j = 0; j < 8; ++j) s[j] = -3.0e38f;
            st[t] = s;
        }

        // ---- online softmax (row stats via half-wave xor shuffles) ---------
        float rm[8];
        for (int j = 0; j < 8; ++j) rm[j] = fmaxf(st[0][j], st[1][j]);
        for (int xm = 1; xm < 16; xm <<= 1)
            for (int j = 0; j < 8; ++j)
                rm[j] = fmaxf(rm[j], __shfl_xor(rm[j], xm, 32));
        float corr[8];
        for (int j = 0; j < 8; ++j) {
            float nm = fmaxf(mstat[j], rm[j]);
            corr[j]  = __expf(mstat[j] - nm);
            mstat[j] = nm;
        }
        float rs[8];
        for (int j = 0; j < 8; ++j) {
            st[0][j] = __expf(st[0][j] - mstat[j]);
            st[1][j] = __expf(st[1][j] - mstat[j]);
            rs[j]    = st[0][j] + st[1][j];
        }
        for (int xm = 1; xm < 16; xm <<= 1)
            for (int j = 0; j < 8; ++j)
                rs[j] += __shfl_xor(rs[j], xm, 32);
        for (int j = 0; j < 8; ++j) {
            lstat[j] = lstat[j] * corr[j] + rs[j];
            for (int d = 0; d < 4; ++d) acc[d][j] *= corr[j];
        }

        // ---- P: C-layout -> A-layout via per-wave LDS tile -----------------
        for (int t = 0; t < 2; ++t)
            for (int j = 0; j < 8; ++j)
                plds[wave][j + 8 * lh][t * 16 + ln] = (__bf16)st[t][j];
        asm volatile("s_wait_dscnt 0" ::: "memory");
        V16 ap;
        ap.h[0] = *(const v8bf*)(&plds[wave][ln][lh * 8]);
        ap.h[1] = *(const v8bf*)(&plds[wave][ln][16 + lh * 8]);

        // ---- acc += P . V  (v transposed in LDS -> contiguous B frags) -----
        for (int dt = 0; dt < 4; ++dt) {
            const __bf16* vp = &vbufs[cur][dt * 16 + ln][lh * 8];
            V16 bv;
            bv.h[0] = *(const v8bf*)(vp);
            bv.h[1] = *(const v8bf*)(vp + 16);
            acc[dt] = wmma_bf16(ap.v, bv.v, acc[dt]);
        }
        __syncthreads();              // protect buffer reuse by next fill
    }

    // ---- normalize and store [b][n][h*64+d] as bf16 ------------------------
    for (int dt = 0; dt < 4; ++dt)
        for (int j = 0; j < 8; ++j) {
            int row = mbase + j + 8 * lh;
            int d   = dt * 16 + ln;
            float o = acc[dt][j] / lstat[j];
            attn_out[((size_t)b * NQ + row) * INNER + h * DH + d] = (__bf16)o;
        }
}

// ---------------------------------------------------------------------------
// Output projection: attn[8192][768] x Wout_t[768][768] + bias -> fp32 out
// ---------------------------------------------------------------------------
__global__ __launch_bounds__(256)
void out_gemm_kernel(const __bf16* __restrict__ ab,
                     const __bf16* __restrict__ wt,
                     const float* __restrict__ bias,
                     float* __restrict__ out) {
    const int wave = threadIdx.x >> 5;
    const int lane = threadIdx.x & 31;
    const int ln   = lane & 15;
    const int lh   = lane >> 4;
    const int m0   = blockIdx.y * 128 + wave * 16;
    const int n0   = blockIdx.x * 64;

    v8f acc[4] = {};
    const __bf16* ar = ab + (size_t)(m0 + ln) * INNER;
    for (int kk = 0; kk < INNER; kk += 32) {
        if (kk + 32 < INNER) __builtin_prefetch(ar + kk + 32, 0, 0);
        V16 a;
        a.h[0] = *(const v8bf*)(ar + kk + lh * 8);
        a.h[1] = *(const v8bf*)(ar + kk + 16 + lh * 8);
        for (int nt = 0; nt < 4; ++nt) {
            const __bf16* wr = wt + (size_t)(n0 + nt * 16 + ln) * INNER + kk + lh * 8;
            V16 b;
            b.h[0] = *(const v8bf*)(wr);
            b.h[1] = *(const v8bf*)(wr + 16);
            acc[nt] = wmma_bf16(a.v, b.v, acc[nt]);
        }
    }
    for (int nt = 0; nt < 4; ++nt) {
        int col = n0 + nt * 16 + ln;
        float bv = bias[col];
        for (int j = 0; j < 8; ++j) {
            int m = m0 + j + 8 * lh;
            out[(size_t)m * DIM + col] = acc[nt][j] + bv;
        }
    }
}

// ---------------------------------------------------------------------------
// Launch
// ---------------------------------------------------------------------------
extern "C" void kernel_launch(void* const* d_in, const int* in_sizes, int n_in,
                              void* d_out, int out_size, void* d_ws, size_t ws_size,
                              hipStream_t stream) {
    const float* x    = (const float*)d_in[0];   // [4,2048,768]
    const float* Wqkv = (const float*)d_in[1];   // [768,2304]
    const float* pk   = (const float*)d_in[2];   // [1,12,8,64]
    const float* pv   = (const float*)d_in[3];   // [1,12,8,64]
    const float* Wout = (const float*)d_in[4];   // [768,768]
    const float* bout = (const float*)d_in[5];   // [768]
    float*       out  = (float*)d_out;           // [4,2048,768]

    char* ws = (char*)d_ws;
    size_t off = 0;
    auto carve = [&](size_t bytes) {
        void* p = ws + off;
        off += (bytes + 255) & ~(size_t)255;
        return p;
    };
    __bf16* xb    = (__bf16*)carve((size_t)MROWS * DIM * 2);
    __bf16* wqkvt = (__bf16*)carve((size_t)QKV_COLS * DIM * 2);
    __bf16* woutt = (__bf16*)carve((size_t)INNER * DIM * 2);
    __bf16* qs    = (__bf16*)carve((size_t)BB * HEADS * NQ * DH * 2);
    __bf16* kb    = (__bf16*)carve((size_t)BB * HEADS * NKPAD * DH * 2);
    __bf16* vt    = (__bf16*)carve((size_t)BB * HEADS * DH * NKPAD * 2);
    __bf16* attnb = (__bf16*)carve((size_t)MROWS * INNER * 2);
    if (off > ws_size) return;

    cvt_copy_kernel<<<4096, 256, 0, stream>>>(x, xb, MROWS * DIM);
    cvt_transpose_kernel<<<(DIM * QKV_COLS + 255) / 256, 256, 0, stream>>>(
        Wqkv, wqkvt, DIM, QKV_COLS);
    cvt_transpose_kernel<<<(DIM * INNER + 255) / 256, 256, 0, stream>>>(
        Wout, woutt, INNER, DIM);

    qkv_gemm_kernel<<<dim3(QKV_COLS / 64, MROWS / 128), 256, 0, stream>>>(
        xb, wqkvt, qs, kb, vt);
    prompt_fill_kernel<<<BB * HEADS, 256, 0, stream>>>(pk, pv, kb, vt);

    attn_kernel<<<dim3(BB * HEADS, NQ / 128), 256, 0, stream>>>(qs, kb, vt, attnb);

    out_gemm_kernel<<<dim3(DIM / 64, MROWS / 128), 256, 0, stream>>>(
        attnb, woutt, bout, out);
}
